// DecoderBlock_50422916055539
// MI455X (gfx1250) — compile-verified, matrix operand order checked by reading, not run on hardware
//
#include <hip/hip_runtime.h>
#include <hip/hip_bf16.h>
#include <math.h>
#include <stdint.h>

// ---------------- problem dims ----------------
constexpr int CB = 2, CS = 2048, CD = 1024, CH = 16, CHD = 64;
constexpr int CE = 8, CK = 2, CF = 2048;
constexpr int CT = CB * CS;      // 4096 tokens
constexpr int CA = CT * CK;      // 8192 routed assignments (always exactly T*K)

typedef __bf16 bf16_t;
typedef __attribute__((ext_vector_type(16))) __bf16 v16bf;
typedef __attribute__((ext_vector_type(8)))  __bf16 v8bf;
typedef __attribute__((ext_vector_type(8)))  float  v8f;

// ---------------- small helpers ----------------
__device__ __forceinline__ bf16_t f2bf(float f){
  union { __hip_bfloat16 h; bf16_t b; } u; u.h = __float2bfloat16(f); return u.b;
}
__device__ __forceinline__ float bf2f(bf16_t b){
  union { bf16_t b; __hip_bfloat16 h; } u; u.b = b; return __bfloat162float(u.h);
}
__device__ __forceinline__ float geluf(float x){              // exact gelu (approximate=False)
  return 0.5f * x * (1.0f + erff(x * 0.70710678118654752440f));
}
__device__ __forceinline__ v8f wmma_bf16(v16bf a, v16bf b, v8f c){
  return __builtin_amdgcn_wmma_f32_16x16x32_bf16(false, a, false, b, (short)0, c, false, false);
}

// 16x32 bf16 fragment (ISA 7.12.2): lane owns row (A) / col (B, from N-major Bt);
// k-half by lane>=16. Two contiguous 16-byte chunks -> global/ds _load_b128.
__device__ __forceinline__ v16bf load_frag_row(const bf16_t* rowbase, int lane){
  const int kh = (lane >> 4) * 8;
  const v8bf lo = *reinterpret_cast<const v8bf*>(rowbase + kh);
  const v8bf hi = *reinterpret_cast<const v8bf*>(rowbase + 16 + kh);
  v16bf r;
#pragma unroll
  for (int i = 0; i < 8; ++i){ r[i] = lo[i]; r[8 + i] = hi[i]; }
  return r;
}

// ---------------- weight transpose + bf16 convert: W[K,N] f32 -> Wt[N,K] bf16 ----------------
__global__ __launch_bounds__(256) void transpose_to_bf16(const float* __restrict__ W,
                                                         bf16_t* __restrict__ Wt,
                                                         int Kd, int N,
                                                         long strideW, long strideWt){
  __shared__ bf16_t tile[32][33];            // +1 pad: no LDS bank conflicts
  const float* Wz = W + (size_t)blockIdx.z * strideW;
  bf16_t* Wtz = Wt + (size_t)blockIdx.z * strideWt;
  const int kt = blockIdx.y * 32, nt = blockIdx.x * 32;
  const int tx = threadIdx.x & 31, ty = threadIdx.x >> 5;
#pragma unroll
  for (int r = ty; r < 32; r += 8)
    tile[r][tx] = f2bf(Wz[(size_t)(kt + r) * N + nt + tx]);
  __syncthreads();
#pragma unroll
  for (int r = ty; r < 32; r += 8)
    Wtz[(size_t)(nt + r) * Kd + kt + tx] = tile[tx][r];
}

// ---------------- RMSNorm: fp32 in -> bf16 out ----------------
__global__ __launch_bounds__(256) void rmsnorm_kernel(const float* __restrict__ x,
                                                      const float* __restrict__ scale,
                                                      bf16_t* __restrict__ out){
  const int t = blockIdx.x;
  const float* xr = x + (size_t)t * CD;
  __shared__ float red[8];
  float s = 0.f;
  for (int i = threadIdx.x; i < CD; i += 256){ float v = xr[i]; s += v * v; }
#pragma unroll
  for (int off = 16; off; off >>= 1) s += __shfl_xor(s, off, 32);
  if ((threadIdx.x & 31) == 0) red[threadIdx.x >> 5] = s;
  __syncthreads();
  if (threadIdx.x < 8){
    float v = red[threadIdx.x];
#pragma unroll
    for (int off = 4; off; off >>= 1) v += __shfl_xor(v, off, 8);
    if (threadIdx.x == 0) red[0] = v;
  }
  __syncthreads();
  const float rn = rsqrtf(red[0] / (float)CD + 1e-8f);
  bf16_t* orow = out + (size_t)t * CD;
  for (int i = threadIdx.x; i < CD; i += 256) orow[i] = f2bf(xr[i] * rn * scale[i]);
}

// ---------------- generic bf16 WMMA GEMM (B pre-transposed to [N,K] bf16) ----------------
// block: 256 thr = 8 waves; tile 64(M) x 32(N); wave = one 16x16 C tile.
// EPI: 0 = bf16(acc+bias), 1 = bf16(gelu(acc+bias)), 2 = f32(acc+bias)+resid, 3 = f32(acc+bias)
template<int EPI>
__global__ __launch_bounds__(256) void gemm_bf16(const bf16_t* __restrict__ A,
                                                 const bf16_t* __restrict__ Bt,
                                                 const float* __restrict__ bias,
                                                 const float* __restrict__ resid,
                                                 float* __restrict__ outF,
                                                 bf16_t* __restrict__ outB,
                                                 int M, int N, int Kd){
  const int lane = threadIdx.x & 31;
  const int wave = threadIdx.x >> 5;
  const int row0 = blockIdx.y * 64 + (wave >> 1) * 16;
  const int col0 = blockIdx.x * 32 + (wave & 1) * 16;
  const bf16_t* arow = A  + (size_t)(row0 + (lane & 15)) * Kd;
  const bf16_t* brow = Bt + (size_t)(col0 + (lane & 15)) * Kd;
  v8f acc = {};
  for (int k = 0; k < Kd; k += 32){
    __builtin_prefetch(brow + k + 256, 0, 1);     // global_prefetch_b8 on weight stream
    v16bf a = load_frag_row(arow + k, lane);
    v16bf b = load_frag_row(brow + k, lane);
    acc = wmma_bf16(a, b, acc);
  }
  const int col = col0 + (lane & 15);
  const float bv = bias[col];
  const int rb = row0 + (lane >> 4) * 8;
#pragma unroll
  for (int v = 0; v < 8; ++v){
    const size_t idx = (size_t)(rb + v) * N + col;
    const float val = acc[v] + bv;
    if (EPI == 0)      outB[idx] = f2bf(val);
    else if (EPI == 1) outB[idx] = f2bf(geluf(val));
    else if (EPI == 2) outF[idx] = val + resid[idx];
    else               outF[idx] = val;
  }
}

// ---------------- RoPE in place on bf16 q/k ----------------
__global__ __launch_bounds__(256) void rope_kernel(bf16_t* __restrict__ qk){
  const int idx = blockIdx.x * 256 + threadIdx.x;        // CT*CH*32 threads
  const int i = idx & 31;
  const int rest = idx >> 5;
  const int h = rest % CH;
  const int t = rest / CH;
  const int s = t % CS;
  const float inv = expf(-9.210340371976184f * ((float)i / 32.0f)); // 10000^(-2i/64)
  const float ang = (float)s * inv;
  const float c = cosf(ang), sn = sinf(ang);
  const size_t base = (size_t)t * CD + h * CHD;
  const float x1 = bf2f(qk[base + i]);
  const float x2 = bf2f(qk[base + i + 32]);
  qk[base + i]      = f2bf(x1 * c - x2 * sn);
  qk[base + i + 32] = f2bf(x2 * c + x1 * sn);
}

// ---------------- flash attention: 1 wave per (b,h,qtile16) ----------------
__global__ __launch_bounds__(32) void attn_kernel(const bf16_t* __restrict__ qb,
                                                  const bf16_t* __restrict__ kb,
                                                  const bf16_t* __restrict__ vb,
                                                  bf16_t* __restrict__ ob){
  const int qt = blockIdx.x, h = blockIdx.y, b = blockIdx.z;
  const int lane = threadIdx.x;
  const int q0 = qt * 16;
  const size_t tok0 = (size_t)b * CS;
  __shared__ bf16_t pTile[16 * 32];
  __shared__ bf16_t vTile[32 * 64];

  const bf16_t* qptr = qb + (tok0 + q0 + (lane & 15)) * CD + h * CHD;
  const v16bf qa0 = load_frag_row(qptr, lane);
  const v16bf qa1 = load_frag_row(qptr + 32, lane);

  v8f o0 = {}, o1 = {}, o2 = {}, o3 = {};
  float m[8], l[8];
#pragma unroll
  for (int v = 0; v < 8; ++v){ m[v] = -INFINITY; l[v] = 0.f; }

  const int n = lane & 15;
  const int rb = (lane >> 4) * 8;
  const int kend = q0 + 16;                   // causal

  for (int kk = 0; kk < kend; kk += 32){
    // ---- scores: Q(16x64) . K^T(64x32) as two 16x16 C frags ----
    v8f s0 = {}, s1 = {};
    {
      const bf16_t* kp0 = kb + (tok0 + kk + n) * CD + h * CHD;      // key = kk+n
      const bf16_t* kp1 = kp0 + (size_t)16 * CD;                    // key = kk+16+n
      v16bf b00 = load_frag_row(kp0, lane), b01 = load_frag_row(kp0 + 32, lane);
      v16bf b10 = load_frag_row(kp1, lane), b11 = load_frag_row(kp1 + 32, lane);
      s0 = wmma_bf16(qa0, b00, s0); s0 = wmma_bf16(qa1, b01, s0);
      s1 = wmma_bf16(qa0, b10, s1); s1 = wmma_bf16(qa1, b11, s1);
    }
    // ---- kick off async V-tile copy global->LDS (one row per lane, 8x b128) ----
    __syncthreads();   // previous iteration's LDS reads are done
    {
      const bf16_t* vp = vb + (tok0 + kk + lane) * CD + h * CHD;
#pragma unroll
      for (int i = 0; i < 8; ++i){
        unsigned ldsoff = (unsigned)(uintptr_t)(vTile + lane * 64 + i * 8);
        unsigned long long ga = (unsigned long long)(uintptr_t)(vp + i * 8);
        asm volatile("global_load_async_to_lds_b128 %0, %1, off"
                     :: "v"(ldsoff), "v"(ga) : "memory");
      }
    }
    // ---- mask + online softmax (row stats reduced across 16-lane halves) ----
#pragma unroll
    for (int v = 0; v < 8; ++v){
      const int qrow = q0 + v + rb;
      float e0 = s0[v] * 0.125f; if (kk + n      > qrow) e0 = -INFINITY;
      float e1 = s1[v] * 0.125f; if (kk + 16 + n > qrow) e1 = -INFINITY;
      float mx = fmaxf(e0, e1);
#pragma unroll
      for (int off = 1; off < 16; off <<= 1) mx = fmaxf(mx, __shfl_xor(mx, off, 32));
      const float mnew = fmaxf(m[v], mx);
      const float scl = __expf(m[v] - mnew);
      const float p0 = __expf(e0 - mnew);
      const float p1 = __expf(e1 - mnew);
      float rs = p0 + p1;
#pragma unroll
      for (int off = 1; off < 16; off <<= 1) rs += __shfl_xor(rs, off, 32);
      l[v] = l[v] * scl + rs;
      m[v] = mnew;
      o0[v] *= scl; o1[v] *= scl; o2[v] *= scl; o3[v] *= scl;
      s0[v] = p0; s1[v] = p1;
    }
    // ---- stage P (C-layout -> A-layout) through LDS ----
#pragma unroll
    for (int v = 0; v < 8; ++v){
      const int r = v + rb;
      pTile[r * 32 + n]      = f2bf(s0[v]);
      pTile[r * 32 + 16 + n] = f2bf(s1[v]);
    }
    asm volatile("s_wait_asynccnt 0x0" ::: "memory");   // V tile resident in LDS
    __syncthreads();
    // ---- O += P(16x32) . V(32x64) ----
    const v16bf pa = load_frag_row(pTile + (lane & 15) * 32, lane);
    const int kh = (lane >> 4) * 8;
#pragma unroll
    for (int nt = 0; nt < 4; ++nt){
      v16bf vf;
#pragma unroll
      for (int i = 0; i < 8; ++i){
        vf[i]     = vTile[(kh + i)      * 64 + nt * 16 + n];
        vf[8 + i] = vTile[(16 + kh + i) * 64 + nt * 16 + n];
      }
      if (nt == 0) o0 = wmma_bf16(pa, vf, o0);
      else if (nt == 1) o1 = wmma_bf16(pa, vf, o1);
      else if (nt == 2) o2 = wmma_bf16(pa, vf, o2);
      else o3 = wmma_bf16(pa, vf, o3);
    }
  }
  // ---- normalize and write ----
#pragma unroll
  for (int v = 0; v < 8; ++v){
    const float invl = 1.f / l[v];
    bf16_t* orow = ob + (tok0 + q0 + v + rb) * CD + h * CHD;
    orow[n]      = f2bf(o0[v] * invl);
    orow[16 + n] = f2bf(o1[v] * invl);
    orow[32 + n] = f2bf(o2[v] * invl);
    orow[48 + n] = f2bf(o3[v] * invl);
  }
}

// ---------------- gating: logits, top-2, softmax, one-hot mask ----------------
__global__ __launch_bounds__(256) void gate_kernel(const bf16_t* __restrict__ hm,
                                                   const float* __restrict__ gw,
                                                   float* __restrict__ mask_out,
                                                   int* __restrict__ tidx,
                                                   float* __restrict__ tp){
  const int t = blockIdx.x * 256 + threadIdx.x;
  if (t >= CT) return;
  float lg[CE];
#pragma unroll
  for (int e = 0; e < CE; ++e) lg[e] = 0.f;
  const bf16_t* hr = hm + (size_t)t * CD;
  for (int d = 0; d < CD; ++d){
    const float hv = bf2f(hr[d]);
    const float* gr = gw + (size_t)d * CE;
#pragma unroll
    for (int e = 0; e < CE; ++e) lg[e] += hv * gr[e];
  }
  int i0 = 0;
#pragma unroll
  for (int e = 1; e < CE; ++e) if (lg[e] > lg[i0]) i0 = e;
  int i1 = (i0 == 0) ? 1 : 0;
#pragma unroll
  for (int e = 0; e < CE; ++e) if (e != i0 && lg[e] > lg[i1]) i1 = e;
  const float ex = expf(lg[i1] - lg[i0]);
  const float p0 = 1.f / (1.f + ex);
  const float p1 = ex / (1.f + ex);
  tidx[t * 2] = i0; tidx[t * 2 + 1] = i1;
  tp[t * 2] = p0;  tp[t * 2 + 1]  = p1;
#pragma unroll
  for (int e = 0; e < CE; ++e){
    mask_out[((size_t)t * 2 + 0) * CE + e] = (e == i0) ? 1.f : 0.f;
    mask_out[((size_t)t * 2 + 1) * CE + e] = (e == i1) ? 1.f : 0.f;
  }
}

// ---------------- deterministic per-expert compaction ----------------
__global__ void route_kernel(const int* __restrict__ tidx, const float* __restrict__ tp,
                             int* __restrict__ seg_tok, float* __restrict__ seg_p,
                             int* __restrict__ inv, int* __restrict__ offs){
  __shared__ int cnt[CE];
  const int e = threadIdx.x;
  if (e < CE){
    int c = 0;
    for (int a = 0; a < CA; ++a) c += (tidx[a] == e);
    cnt[e] = c;
  }
  __syncthreads();
  if (e < CE){
    int base = 0;
    for (int j = 0; j < e; ++j) base += cnt[j];
    if (e == 0){
      int s = 0;
      for (int j = 0; j < CE; ++j){ offs[j] = s; s += cnt[j]; }
      offs[CE] = s;
    }
    for (int a = 0; a < CA; ++a){
      if (tidx[a] == e){
        seg_tok[base] = a >> 1;
        seg_p[base]   = tp[a];
        inv[a]        = base;
        ++base;
      }
    }
  }
}

// ---------------- grouped MoE GEMM 1: h1 = gelu(hm . ew1[e] + eb1[e]) ----------------
__global__ __launch_bounds__(256) void moe_ffn1(const bf16_t* __restrict__ hm,
                                                const bf16_t* __restrict__ ew1t, // [E][F][D] bf16
                                                const float* __restrict__ eb1,
                                                const int* __restrict__ seg_tok,
                                                const int* __restrict__ offs,
                                                bf16_t* __restrict__ h1){
  const int e = blockIdx.z;
  const int seg_end = offs[e + 1];
  const int r0 = offs[e] + blockIdx.y * 16;
  if (r0 >= seg_end) return;                       // uniform early exit (EXEC stays full)
  const int lane = threadIdx.x & 31;
  const int wave = threadIdx.x >> 5;
  const int col0 = blockIdx.x * 128 + wave * 16;
  int rl = r0 + (lane & 15);
  if (rl >= seg_end) rl = seg_end - 1;             // clamp row gather inside segment
  const bf16_t* arow = hm + (size_t)seg_tok[rl] * CD;
  const bf16_t* brow = ew1t + (size_t)e * CD * CF + (size_t)(col0 + (lane & 15)) * CD;
  v8f acc = {};
  for (int k = 0; k < CD; k += 32){
    __builtin_prefetch(brow + k + 256, 0, 1);
    v16bf a = load_frag_row(arow + k, lane);
    v16bf b = load_frag_row(brow + k, lane);
    acc = wmma_bf16(a, b, acc);
  }
  const int col = col0 + (lane & 15);
  const float bias = eb1[e * CF + col];
  const int rbv = r0 + (lane >> 4) * 8;
#pragma unroll
  for (int v = 0; v < 8; ++v){
    const int r = rbv + v;
    if (r < seg_end) h1[(size_t)r * CF + col] = f2bf(geluf(acc[v] + bias));
  }
}

// ---------------- grouped MoE GEMM 2: ey = p * (h1 . ew2[e] + eb2[e]) ----------------
__global__ __launch_bounds__(256) void moe_ffn2(const bf16_t* __restrict__ h1,
                                                const bf16_t* __restrict__ ew2t, // [E][D][F] bf16
                                                const float* __restrict__ eb2,
                                                const float* __restrict__ seg_p,
                                                const int* __restrict__ offs,
                                                float* __restrict__ ey){
  const int e = blockIdx.z;
  const int seg_end = offs[e + 1];
  const int r0 = offs[e] + blockIdx.y * 16;
  if (r0 >= seg_end) return;
  const int lane = threadIdx.x & 31;
  const int wave = threadIdx.x >> 5;
  const int col0 = blockIdx.x * 128 + wave * 16;
  int rl = r0 + (lane & 15);
  if (rl >= seg_end) rl = seg_end - 1;
  const bf16_t* arow = h1 + (size_t)rl * CF;
  const bf16_t* brow = ew2t + (size_t)e * CF * CD + (size_t)(col0 + (lane & 15)) * CF;
  v8f acc = {};
  for (int k = 0; k < CF; k += 32){
    __builtin_prefetch(brow + k + 256, 0, 1);
    v16bf a = load_frag_row(arow + k, lane);
    v16bf b = load_frag_row(brow + k, lane);
    acc = wmma_bf16(a, b, acc);
  }
  const int col = col0 + (lane & 15);
  const float bias = eb2[e * CD + col];
  const int rbv = r0 + (lane >> 4) * 8;
#pragma unroll
  for (int v = 0; v < 8; ++v){
    const int r = rbv + v;
    if (r < seg_end) ey[(size_t)r * CD + col] = seg_p[r] * (acc[v] + bias);
  }
}

// ---------------- final combine: x1 + shared + top2 expert rows ----------------
__global__ __launch_bounds__(256) void final_kernel(const float* __restrict__ x1,
                                                    const float* __restrict__ sh2,
                                                    const float* __restrict__ ey,
                                                    const int* __restrict__ inv,
                                                    float* __restrict__ out){
  const size_t idx = (size_t)blockIdx.x * 256 + threadIdx.x;
  if (idx >= (size_t)CT * CD) return;
  const int t = (int)(idx / CD);
  const int c = (int)(idx % CD);
  const int r0 = inv[2 * t], r1 = inv[2 * t + 1];
  out[idx] = x1[idx] + sh2[idx] + ey[(size_t)r0 * CD + c] + ey[(size_t)r1 * CD + c];
}

// ---------------- host launcher ----------------
extern "C" void kernel_launch(void* const* d_in, const int* in_sizes, int n_in,
                              void* d_out, int out_size, void* d_ws, size_t ws_size,
                              hipStream_t stream){
  (void)in_sizes; (void)n_in; (void)out_size; (void)ws_size;
  const float* x   = (const float*)d_in[0];
  /* d_in[1] = causal mask (bool) - computed analytically */
  const float* ans = (const float*)d_in[2];
  const float* wq  = (const float*)d_in[3];
  const float* bq  = (const float*)d_in[4];
  const float* wk  = (const float*)d_in[5];
  const float* bk  = (const float*)d_in[6];
  const float* wv  = (const float*)d_in[7];
  const float* bv  = (const float*)d_in[8];
  const float* wo  = (const float*)d_in[9];
  const float* bo  = (const float*)d_in[10];
  const float* mns = (const float*)d_in[11];
  const float* gw  = (const float*)d_in[12];
  const float* ew1 = (const float*)d_in[13];
  const float* eb1 = (const float*)d_in[14];
  const float* ew2 = (const float*)d_in[15];
  const float* eb2 = (const float*)d_in[16];
  const float* sw1 = (const float*)d_in[17];
  const float* sb1 = (const float*)d_in[18];
  const float* sw2 = (const float*)d_in[19];
  const float* sb2 = (const float*)d_in[20];

  char* p = (char*)d_ws;
  auto carve = [&](size_t bytes) -> char* {
    char* q = p; p += (bytes + 255) & ~(size_t)255; return q;
  };
  bf16_t* hb   = (bf16_t*)carve((size_t)CT * CD * 2);
  bf16_t* qb   = (bf16_t*)carve((size_t)CT * CD * 2);
  bf16_t* kb   = (bf16_t*)carve((size_t)CT * CD * 2);
  bf16_t* vb   = (bf16_t*)carve((size_t)CT * CD * 2);
  bf16_t* ob   = (bf16_t*)carve((size_t)CT * CD * 2);
  float*  x1   = (float*) carve((size_t)CT * CD * 4);
  bf16_t* hmb  = (bf16_t*)carve((size_t)CT * CD * 2);
  bf16_t* h1s  = (bf16_t*)carve((size_t)CT * CF * 2);
  float*  sh2  = (float*) carve((size_t)CT * CD * 4);
  bf16_t* h1e  = (bf16_t*)carve((size_t)CA * CF * 2);
  float*  ey   = (float*) carve((size_t)CA * CD * 4);
  // transposed bf16 weights
  bf16_t* wqT  = (bf16_t*)carve((size_t)CD * CD * 2);
  bf16_t* wkT  = (bf16_t*)carve((size_t)CD * CD * 2);
  bf16_t* wvT  = (bf16_t*)carve((size_t)CD * CD * 2);
  bf16_t* woT  = (bf16_t*)carve((size_t)CD * CD * 2);
  bf16_t* sw1T = (bf16_t*)carve((size_t)CD * CF * 2);
  bf16_t* sw2T = (bf16_t*)carve((size_t)CF * CD * 2);
  bf16_t* ew1T = (bf16_t*)carve((size_t)CE * CD * CF * 2);
  bf16_t* ew2T = (bf16_t*)carve((size_t)CE * CF * CD * 2);
  int*    tidx = (int*)   carve((size_t)CA * 4);
  float*  tp   = (float*) carve((size_t)CA * 4);
  int*    stok = (int*)   carve((size_t)CA * 4);
  float*  sp   = (float*) carve((size_t)CA * 4);
  int*    inv  = (int*)   carve((size_t)CA * 4);
  int*    offs = (int*)   carve((size_t)(CE + 1) * 4);

  float* outX    = (float*)d_out;
  float* maskOut = outX + (size_t)CT * CD;

  // ---- weight preprocessing: f32 [K,N] -> bf16 [N,K] ----
  transpose_to_bf16<<<dim3(CD/32, CD/32, 1),  256, 0, stream>>>(wq,  wqT,  CD, CD, 0, 0);
  transpose_to_bf16<<<dim3(CD/32, CD/32, 1),  256, 0, stream>>>(wk,  wkT,  CD, CD, 0, 0);
  transpose_to_bf16<<<dim3(CD/32, CD/32, 1),  256, 0, stream>>>(wv,  wvT,  CD, CD, 0, 0);
  transpose_to_bf16<<<dim3(CD/32, CD/32, 1),  256, 0, stream>>>(wo,  woT,  CD, CD, 0, 0);
  transpose_to_bf16<<<dim3(CF/32, CD/32, 1),  256, 0, stream>>>(sw1, sw1T, CD, CF, 0, 0);
  transpose_to_bf16<<<dim3(CD/32, CF/32, 1),  256, 0, stream>>>(sw2, sw2T, CF, CD, 0, 0);
  transpose_to_bf16<<<dim3(CF/32, CD/32, CE), 256, 0, stream>>>(ew1, ew1T, CD, CF,
                                                                (long)CD*CF, (long)CD*CF);
  transpose_to_bf16<<<dim3(CD/32, CF/32, CE), 256, 0, stream>>>(ew2, ew2T, CF, CD,
                                                                (long)CF*CD, (long)CF*CD);

  // ---- attention sublayer ----
  rmsnorm_kernel<<<CT, 256, 0, stream>>>(x, ans, hb);
  gemm_bf16<0><<<dim3(CD / 32, CT / 64), 256, 0, stream>>>(hb, wqT, bq, nullptr, nullptr, qb, CT, CD, CD);
  gemm_bf16<0><<<dim3(CD / 32, CT / 64), 256, 0, stream>>>(hb, wkT, bk, nullptr, nullptr, kb, CT, CD, CD);
  gemm_bf16<0><<<dim3(CD / 32, CT / 64), 256, 0, stream>>>(hb, wvT, bv, nullptr, nullptr, vb, CT, CD, CD);
  rope_kernel<<<(CT * CH * 32) / 256, 256, 0, stream>>>(qb);
  rope_kernel<<<(CT * CH * 32) / 256, 256, 0, stream>>>(kb);
  attn_kernel<<<dim3(CS / 16, CH, CB), 32, 0, stream>>>(qb, kb, vb, ob);
  gemm_bf16<2><<<dim3(CD / 32, CT / 64), 256, 0, stream>>>(ob, woT, bo, x, x1, nullptr, CT, CD, CD);

  // ---- MoE sublayer ----
  rmsnorm_kernel<<<CT, 256, 0, stream>>>(x1, mns, hmb);
  gate_kernel<<<CT / 256, 256, 0, stream>>>(hmb, gw, maskOut, tidx, tp);
  route_kernel<<<1, 32, 0, stream>>>(tidx, tp, stok, sp, inv, offs);
  moe_ffn1<<<dim3(CF / 128, CT / 16, CE), 256, 0, stream>>>(hmb, ew1T, eb1, stok, offs, h1e);
  moe_ffn2<<<dim3(CD / 128, CT / 16, CE), 256, 0, stream>>>(h1e, ew2T, eb2, sp, offs, ey);
  gemm_bf16<1><<<dim3(CF / 32, CT / 64), 256, 0, stream>>>(hmb, sw1T, sb1, nullptr, nullptr, h1s, CT, CF, CD);
  gemm_bf16<3><<<dim3(CD / 32, CT / 64), 256, 0, stream>>>(h1s, sw2T, sb2, nullptr, sh2, nullptr, CT, CD, CF);
  final_kernel<<<((size_t)CT * CD + 255) / 256, 256, 0, stream>>>(x1, sh2, ey, inv, outX);
}
